// TemporalMoEBlock_85950885527617
// MI455X (gfx1250) — compile-verified, matrix operand order checked by reading, not run on hardware
//
#include <hip/hip_runtime.h>

// ---------------------------------------------------------------------------
// TemporalMoEBlock for MI455X (gfx1250, wave32, WMMA).
// All big contractions use v_wmma_f32_16x16x32_bf16 (bf16 in, fp32 accum).
// B=2 T=512 D=1024 H=16 DH=64 E=8 K=2 F=4096.
// ---------------------------------------------------------------------------

typedef unsigned short u16;
typedef __attribute__((ext_vector_type(16))) __bf16 v16bf;
typedef __attribute__((ext_vector_type(8)))  __bf16 v8bf;
typedef __attribute__((ext_vector_type(8)))  float  v8f;

#define NB 1024   // B*T tokens
#define DD 1024
#define TT 512
#define HH 16
#define DH 64
#define EE 8
#define FF 4096

__device__ __forceinline__ u16 f2bf(float f) {
  unsigned u = __float_as_uint(f);
  unsigned r = u + 0x7FFFu + ((u >> 16) & 1u);   // round-to-nearest-even
  return (u16)(r >> 16);
}

__device__ __forceinline__ v8f vzero8() {
  v8f z; 
#pragma unroll
  for (int i = 0; i < 8; ++i) z[i] = 0.0f;
  return z;
}

// 16-element bf16 fragment load as two 16B vectors (16B alignment guaranteed).
__device__ __forceinline__ v16bf ld_frag16(const u16* p) {
  union { v16bf v; v8bf h[2]; } u;
  u.h[0] = *reinterpret_cast<const v8bf*>(p);
  u.h[1] = *reinterpret_cast<const v8bf*>(p + 8);
  return u.v;
}

__device__ __forceinline__ v8f wmma_bf16(v16bf a, v16bf b, v8f c) {
  return __builtin_amdgcn_wmma_f32_16x16x32_bf16(
      /*neg_a=*/false, a, /*neg_b=*/false, b,
      /*c_mod=*/(short)0, c, /*reuse_a=*/false, /*reuse_b=*/false);
}

__device__ __forceinline__ float gelu_tanh(float x) {
  float x3 = x * x * x;
  return 0.5f * x * (1.0f + tanhf(0.7978845608f * (x + 0.044715f * x3)));
}

// Stage a 32(K) x 64(N) fp32 weight tile, transposed+converted to bf16:
// Bs[n*40 + k].  W must point at element (k0, n0).
__device__ __forceinline__ void stage_wtile_f32(const float* __restrict__ W, int ldb,
                                                u16* __restrict__ Bs, int tid) {
#pragma unroll
  for (int i = 0; i < 16; ++i) {
    int e  = i * 128 + tid;
    int kk = e >> 6;
    int nn = e & 63;
    Bs[nn * 40 + kk] = f2bf(W[(size_t)kk * ldb + nn]);
  }
}

// Stage a 32(K=s) x 64(N=d) bf16 V tile transposed: Bs[d*40 + s].
__device__ __forceinline__ void stage_vtile_bf(const u16* __restrict__ V,
                                               u16* __restrict__ Bs, int tid) {
#pragma unroll
  for (int i = 0; i < 16; ++i) {
    int e  = i * 128 + tid;
    int kk = e >> 6;
    int nn = e & 63;
    Bs[nn * 40 + kk] = V[(size_t)kk * 64 + nn];
  }
}

// One 32-K WMMA step with B from LDS (Bs pitch 40).
__device__ __forceinline__ void wmma_step_lds(const u16* a0, const u16* a1,
                                              const u16* __restrict__ Bs,
                                              int wn, int lm, int kh,
                                              v8f acc[2][2]) {
  v16bf af0 = ld_frag16(a0);
  v16bf af1 = ld_frag16(a1);
#pragma unroll
  for (int tn = 0; tn < 2; ++tn) {
    v16bf bf_ = ld_frag16(&Bs[(wn + tn * 16 + lm) * 40 + kh]);
    acc[0][tn] = wmma_bf16(af0, bf_, acc[0][tn]);
    acc[1][tn] = wmma_bf16(af1, bf_, acc[1][tn]);
  }
}

// ---------------------------------------------------------------------------
// LayerNorm: one block per token row of 1024. Writes bf16 (and optional fp32).
// ---------------------------------------------------------------------------
__global__ void ln_kernel(const float* __restrict__ x, const float* __restrict__ gs,
                          const float* __restrict__ gb, u16* __restrict__ obf,
                          float* __restrict__ of) {
  const int row = blockIdx.x, tid = threadIdx.x;
  const float* p = x + (size_t)row * DD;
  float v[4], s = 0.f, s2 = 0.f;
#pragma unroll
  for (int i = 0; i < 4; ++i) { v[i] = p[tid + i * 256]; s += v[i]; s2 += v[i] * v[i]; }
  __shared__ float r1[256], r2[256];
  r1[tid] = s; r2[tid] = s2; __syncthreads();
  for (int st = 128; st > 0; st >>= 1) {
    if (tid < st) { r1[tid] += r1[tid + st]; r2[tid] += r2[tid + st]; }
    __syncthreads();
  }
  float mu  = r1[0] * (1.0f / DD);
  float var = r2[0] * (1.0f / DD) - mu * mu;
  float rs  = rsqrtf(var + 1e-5f);
#pragma unroll
  for (int i = 0; i < 4; ++i) {
    int c = tid + i * 256;
    float hv = (v[i] - mu) * rs * gs[c] + gb[c];
    obf[(size_t)row * DD + c] = f2bf(hv);
    if (of) of[(size_t)row * DD + c] = hv;
  }
}

// ---------------------------------------------------------------------------
// QKV GEMM: (1024 x 3072) = h1_bf @ w_qkv + b. Epilogue scatters q(k)(v)
// into (B,H,T,DH) bf16; q pre-scaled by 1/sqrt(DH).
// ---------------------------------------------------------------------------
__global__ void qkv_gemm(const u16* __restrict__ A, const float* __restrict__ W,
                         const float* __restrict__ bias, u16* __restrict__ qbf,
                         u16* __restrict__ kbf, u16* __restrict__ vbf) {
  __shared__ u16 Bs[64 * 40];
  const int tid = threadIdx.x, lane = tid & 31, wave = tid >> 5;
  const int lm = lane & 15, kh = (lane >> 4) * 16;
  const int wm = (wave >> 1) * 32, wn = (wave & 1) * 32;
  const int tileM = blockIdx.y * 64, tileN = blockIdx.x * 64;

  const u16* a0 = A + (size_t)(tileM + wm + lm) * DD + kh;
  const u16* a1 = A + (size_t)(tileM + wm + 16 + lm) * DD + kh;
  v8f acc[2][2] = {{vzero8(), vzero8()}, {vzero8(), vzero8()}};

  for (int k0 = 0; k0 < DD; k0 += 32) {
    __syncthreads();
    stage_wtile_f32(W + (size_t)k0 * (3 * DD) + tileN, 3 * DD, Bs, tid);
    __syncthreads();
    wmma_step_lds(a0 + k0, a1 + k0, Bs, wn, lm, kh, acc);
  }
#pragma unroll
  for (int tm = 0; tm < 2; ++tm)
#pragma unroll
    for (int tn = 0; tn < 2; ++tn)
#pragma unroll
      for (int r = 0; r < 8; ++r) {
        int row = tileM + wm + tm * 16 + (lane >> 4) * 8 + r;
        int col = tileN + wn + tn * 16 + lm;
        float val = acc[tm][tn][r] + bias[col];
        int part = col >> 10, hh = (col & 1023) >> 6, dd = col & 63;
        int b = row >> 9, t = row & 511;
        size_t di = (((size_t)(b * HH + hh) * TT) + t) * DH + dd;
        if (part == 0)      qbf[di] = f2bf(val * 0.125f);
        else if (part == 1) kbf[di] = f2bf(val);
        else                vbf[di] = f2bf(val);
      }
}

// ---------------------------------------------------------------------------
// Scores: per (b,h) 512x512 = q @ k^T + rel_bias. No LDS: both operands
// are K-contiguous bf16 rows in global.
// ---------------------------------------------------------------------------
__global__ void scores_gemm(const u16* __restrict__ qbf, const u16* __restrict__ kbf,
                            const float* __restrict__ rel, float* __restrict__ scores) {
  const int tid = threadIdx.x, lane = tid & 31, wave = tid >> 5;
  const int lm = lane & 15, kh = (lane >> 4) * 16;
  const int wm = (wave >> 1) * 32, wn = (wave & 1) * 32;
  const int bh = blockIdx.z, h = bh & (HH - 1);
  const int tileT = blockIdx.y * 64, tileS = blockIdx.x * 64;
  const size_t base = (size_t)bh * TT * DH;

  const u16* a0 = qbf + base + (size_t)(tileT + wm + lm) * DH + kh;
  const u16* a1 = qbf + base + (size_t)(tileT + wm + 16 + lm) * DH + kh;
  const u16* b0 = kbf + base + (size_t)(tileS + wn + lm) * DH + kh;
  const u16* b1 = kbf + base + (size_t)(tileS + wn + 16 + lm) * DH + kh;

  v8f acc[2][2] = {{vzero8(), vzero8()}, {vzero8(), vzero8()}};
#pragma unroll
  for (int k0 = 0; k0 < DH; k0 += 32) {
    v16bf af0 = ld_frag16(a0 + k0), af1 = ld_frag16(a1 + k0);
    v16bf bf0 = ld_frag16(b0 + k0), bf1 = ld_frag16(b1 + k0);
    acc[0][0] = wmma_bf16(af0, bf0, acc[0][0]);
    acc[0][1] = wmma_bf16(af0, bf1, acc[0][1]);
    acc[1][0] = wmma_bf16(af1, bf0, acc[1][0]);
    acc[1][1] = wmma_bf16(af1, bf1, acc[1][1]);
  }
#pragma unroll
  for (int tm = 0; tm < 2; ++tm)
#pragma unroll
    for (int tn = 0; tn < 2; ++tn)
#pragma unroll
      for (int r = 0; r < 8; ++r) {
        int t = tileT + wm + tm * 16 + (lane >> 4) * 8 + r;
        int s = tileS + wn + tn * 16 + lm;
        float val = acc[tm][tn][r] + rel[h * (2 * TT - 1) + (s - t + TT - 1)];
        scores[((size_t)bh * TT + t) * TT + s] = val;
      }
}

// ---------------------------------------------------------------------------
// Row softmax over 512, output bf16 attention weights.
// ---------------------------------------------------------------------------
__global__ void softmax_kernel(const float* __restrict__ scores, u16* __restrict__ attn) {
  const int row = blockIdx.x, tid = threadIdx.x;
  const float* p = scores + (size_t)row * TT;
  __shared__ float red[128];
  float v[4], m = -1e30f;
#pragma unroll
  for (int i = 0; i < 4; ++i) { v[i] = p[tid + i * 128]; m = fmaxf(m, v[i]); }
  red[tid] = m; __syncthreads();
  for (int st = 64; st > 0; st >>= 1) {
    if (tid < st) red[tid] = fmaxf(red[tid], red[tid + st]);
    __syncthreads();
  }
  m = red[0]; __syncthreads();
  float s = 0.f;
#pragma unroll
  for (int i = 0; i < 4; ++i) { v[i] = __expf(v[i] - m); s += v[i]; }
  red[tid] = s; __syncthreads();
  for (int st = 64; st > 0; st >>= 1) {
    if (tid < st) red[tid] += red[tid + st];
    __syncthreads();
  }
  float inv = 1.0f / red[0];
#pragma unroll
  for (int i = 0; i < 4; ++i) attn[(size_t)row * TT + tid + i * 128] = f2bf(v[i] * inv);
}

// ---------------------------------------------------------------------------
// Context: per (b,h) 512x64 = attn @ v. V staged transposed through LDS.
// ---------------------------------------------------------------------------
__global__ void ctx_gemm(const u16* __restrict__ attnbf, const u16* __restrict__ vbf,
                         u16* __restrict__ ctxbf) {
  __shared__ u16 Bs[64 * 40];
  const int tid = threadIdx.x, lane = tid & 31, wave = tid >> 5;
  const int lm = lane & 15, kh = (lane >> 4) * 16;
  const int wm = (wave >> 1) * 32, wn = (wave & 1) * 32;
  const int bh = blockIdx.y, tileT = blockIdx.x * 64;
  const int b = bh >> 4, h = bh & (HH - 1);

  const u16* a0 = attnbf + ((size_t)bh * TT + tileT + wm + lm) * TT + kh;
  const u16* a1 = attnbf + ((size_t)bh * TT + tileT + wm + 16 + lm) * TT + kh;
  v8f acc[2][2] = {{vzero8(), vzero8()}, {vzero8(), vzero8()}};

  for (int k0 = 0; k0 < TT; k0 += 32) {
    __syncthreads();
    stage_vtile_bf(vbf + ((size_t)bh * TT + k0) * DH, Bs, tid);
    __syncthreads();
    wmma_step_lds(a0 + k0, a1 + k0, Bs, wn, lm, kh, acc);
  }
#pragma unroll
  for (int tm = 0; tm < 2; ++tm)
#pragma unroll
    for (int tn = 0; tn < 2; ++tn)
#pragma unroll
      for (int r = 0; r < 8; ++r) {
        int t = tileT + wm + tm * 16 + (lane >> 4) * 8 + r;
        int d = wn + tn * 16 + lm;
        ctxbf[((size_t)b * TT + t) * DD + h * DH + d] = f2bf(acc[tm][tn][r]);
      }
}

// ---------------------------------------------------------------------------
// Output projection + residual: out = x + ctx @ w_o + b_o  (1024x1024x1024)
// ---------------------------------------------------------------------------
__global__ void outproj_gemm(const u16* __restrict__ A, const float* __restrict__ W,
                             const float* __restrict__ bias, const float* __restrict__ xin,
                             float* __restrict__ out) {
  __shared__ u16 Bs[64 * 40];
  const int tid = threadIdx.x, lane = tid & 31, wave = tid >> 5;
  const int lm = lane & 15, kh = (lane >> 4) * 16;
  const int wm = (wave >> 1) * 32, wn = (wave & 1) * 32;
  const int tileM = blockIdx.y * 64, tileN = blockIdx.x * 64;

  const u16* a0 = A + (size_t)(tileM + wm + lm) * DD + kh;
  const u16* a1 = A + (size_t)(tileM + wm + 16 + lm) * DD + kh;
  v8f acc[2][2] = {{vzero8(), vzero8()}, {vzero8(), vzero8()}};

  for (int k0 = 0; k0 < DD; k0 += 32) {
    __syncthreads();
    stage_wtile_f32(W + (size_t)k0 * DD + tileN, DD, Bs, tid);
    __syncthreads();
    wmma_step_lds(a0 + k0, a1 + k0, Bs, wn, lm, kh, acc);
  }
#pragma unroll
  for (int tm = 0; tm < 2; ++tm)
#pragma unroll
    for (int tn = 0; tn < 2; ++tn)
#pragma unroll
      for (int r = 0; r < 8; ++r) {
        int row = tileM + wm + tm * 16 + (lane >> 4) * 8 + r;
        int col = tileN + wn + tn * 16 + lm;
        size_t ix = (size_t)row * DD + col;
        out[ix] = xin[ix] + acc[tm][tn][r] + bias[col];
      }
}

// ---------------------------------------------------------------------------
// text_state @ w_txt + b_txt   (B x D, small)
// ---------------------------------------------------------------------------
__global__ void txtproj_kernel(const float* __restrict__ ts, const float* __restrict__ W,
                               const float* __restrict__ bias, float* __restrict__ out) {
  int idx = blockIdx.x * 128 + threadIdx.x;       // 2048 outputs
  int b = idx >> 10, n = idx & 1023;
  float acc = bias[n];
  for (int d = 0; d < DD; ++d) acc += ts[b * DD + d] * W[(size_t)d * DD + n];
  out[idx] = acc;
}

// ---------------------------------------------------------------------------
// Router: per-token 8 logits, softmax, top-2 gates (first-index tie rule).
// ---------------------------------------------------------------------------
__global__ void router_kernel(const float* __restrict__ h2f, const float* __restrict__ txt,
                              const float* __restrict__ wr, const float* __restrict__ br,
                              float* __restrict__ probs, float* __restrict__ gatev,
                              int* __restrict__ topidx) {
  const int tok = blockIdx.x, tid = threadIdx.x, b = tok >> 9;
  __shared__ float lg[EE];
  if (tid < EE) {
    float acc = br[tid];
    const float* hp = h2f + (size_t)tok * DD;
    const float* tp = txt + (size_t)b * DD;
    for (int d = 0; d < DD; ++d) acc += (hp[d] + tp[d]) * wr[d * EE + tid];
    lg[tid] = acc;
  }
  __syncthreads();
  if (tid == 0) {
    float m = lg[0];
    for (int e = 1; e < EE; ++e) m = fmaxf(m, lg[e]);
    float pr[EE], s = 0.f;
    for (int e = 0; e < EE; ++e) { pr[e] = __expf(lg[e] - m); s += pr[e]; }
    float inv = 1.0f / s;
    for (int e = 0; e < EE; ++e) { pr[e] *= inv; probs[tok * EE + e] = pr[e]; }
    int i0 = 0; float v0 = pr[0];
    for (int e = 1; e < EE; ++e) if (pr[e] > v0) { v0 = pr[e]; i0 = e; }
    int i1 = -1; float v1 = -1.0f;
    for (int e = 0; e < EE; ++e) if (e != i0 && pr[e] > v1) { v1 = pr[e]; i1 = e; }
    float den = 1.0f / (v0 + v1);
    topidx[tok * 2 + 0] = i0; topidx[tok * 2 + 1] = i1;
    gatev[tok * 2 + 0] = v0 * den; gatev[tok * 2 + 1] = v1 * den;
  }
}

// ---------------------------------------------------------------------------
// Deterministic serial expert-list build: counts, offsets, token lists,
// per-(token,slot) compact row id.
// ---------------------------------------------------------------------------
__global__ void build_lists_kernel(const int* __restrict__ topidx, int* __restrict__ cnt,
                                   int* __restrict__ off, int* __restrict__ list,
                                   int* __restrict__ rowid) {
  if (threadIdx.x != 0) return;
  for (int e = 0; e < EE; ++e) cnt[e] = 0;
  for (int t = 0; t < NB; ++t)
    for (int k = 0; k < 2; ++k) {
      int e = topidx[t * 2 + k];
      rowid[t * 2 + k] = cnt[e];
      list[e * NB + cnt[e]] = t;
      cnt[e]++;
    }
  off[0] = 0;
  for (int e = 0; e < EE; ++e) off[e + 1] = off[e] + cnt[e];
  for (int t = 0; t < NB; ++t)
    for (int k = 0; k < 2; ++k) rowid[t * 2 + k] += off[topidx[t * 2 + k]];
}

// ---------------------------------------------------------------------------
// MoE GEMM1: hid = gelu(gather(h2) @ w1[e] + b1[e]) -> bf16 compact rows.
// ---------------------------------------------------------------------------
__global__ void moe_gemm1(const u16* __restrict__ h2bf, const float* __restrict__ w1,
                          const float* __restrict__ b1, const int* __restrict__ cnt,
                          const int* __restrict__ off, const int* __restrict__ list,
                          u16* __restrict__ hid) {
  __shared__ u16 Bs[64 * 40];
  const int e = blockIdx.z;
  const int mcount = cnt[e];
  const int tileM = blockIdx.y * 64;
  if (tileM >= mcount) return;
  const int tileN = blockIdx.x * 64;
  const int tid = threadIdx.x, lane = tid & 31, wave = tid >> 5;
  const int lm = lane & 15, kh = (lane >> 4) * 16;
  const int wm = (wave >> 1) * 32, wn = (wave & 1) * 32;

  int rl0 = tileM + wm + lm, rl1 = tileM + wm + 16 + lm;
  int t0 = list[e * NB + (rl0 < mcount ? rl0 : mcount - 1)];
  int t1 = list[e * NB + (rl1 < mcount ? rl1 : mcount - 1)];
  const u16* a0 = h2bf + (size_t)t0 * DD + kh;
  const u16* a1 = h2bf + (size_t)t1 * DD + kh;
  const float* W = w1 + (size_t)e * DD * FF;
  v8f acc[2][2] = {{vzero8(), vzero8()}, {vzero8(), vzero8()}};

  for (int k0 = 0; k0 < DD; k0 += 32) {
    __syncthreads();
    stage_wtile_f32(W + (size_t)k0 * FF + tileN, FF, Bs, tid);
    __syncthreads();
    wmma_step_lds(a0 + k0, a1 + k0, Bs, wn, lm, kh, acc);
  }
  const int base = off[e];
#pragma unroll
  for (int tm = 0; tm < 2; ++tm)
#pragma unroll
    for (int tn = 0; tn < 2; ++tn)
#pragma unroll
      for (int r = 0; r < 8; ++r) {
        int row = tileM + wm + tm * 16 + (lane >> 4) * 8 + r;
        int col = tileN + wn + tn * 16 + lm;
        if (row < mcount) {
          float val = acc[tm][tn][r] + b1[e * FF + col];
          hid[(size_t)(base + row) * FF + col] = f2bf(gelu_tanh(val));
        }
      }
}

// ---------------------------------------------------------------------------
// MoE GEMM2: eout = hid @ w2[e] + b2[e]  (compact rows, K=4096)
// ---------------------------------------------------------------------------
__global__ void moe_gemm2(const u16* __restrict__ hid, const float* __restrict__ w2,
                          const float* __restrict__ b2, const int* __restrict__ cnt,
                          const int* __restrict__ off, float* __restrict__ eout) {
  __shared__ u16 Bs[64 * 40];
  const int e = blockIdx.z;
  const int mcount = cnt[e];
  const int tileM = blockIdx.y * 64;
  if (tileM >= mcount) return;
  const int tileN = blockIdx.x * 64;
  const int tid = threadIdx.x, lane = tid & 31, wave = tid >> 5;
  const int lm = lane & 15, kh = (lane >> 4) * 16;
  const int wm = (wave >> 1) * 32, wn = (wave & 1) * 32;
  const int base = off[e];

  int rl0 = tileM + wm + lm, rl1 = tileM + wm + 16 + lm;
  rl0 = rl0 < mcount ? rl0 : mcount - 1;
  rl1 = rl1 < mcount ? rl1 : mcount - 1;
  const u16* a0 = hid + (size_t)(base + rl0) * FF + kh;
  const u16* a1 = hid + (size_t)(base + rl1) * FF + kh;
  const float* W = w2 + (size_t)e * FF * DD;
  v8f acc[2][2] = {{vzero8(), vzero8()}, {vzero8(), vzero8()}};

  for (int k0 = 0; k0 < FF; k0 += 32) {
    __syncthreads();
    stage_wtile_f32(W + (size_t)k0 * DD + tileN, DD, Bs, tid);
    __syncthreads();
    wmma_step_lds(a0 + k0, a1 + k0, Bs, wn, lm, kh, acc);
  }
#pragma unroll
  for (int tm = 0; tm < 2; ++tm)
#pragma unroll
    for (int tn = 0; tn < 2; ++tn)
#pragma unroll
      for (int r = 0; r < 8; ++r) {
        int row = tileM + wm + tm * 16 + (lane >> 4) * 8 + r;
        int col = tileN + wn + tn * 16 + lm;
        if (row < mcount)
          eout[(size_t)(base + row) * DD + col] = acc[tm][tn][r] + b2[e * DD + col];
      }
}

// ---------------------------------------------------------------------------
// Deterministic gated combine: out[t] += g0*eout[row0] + g1*eout[row1]
// ---------------------------------------------------------------------------
__global__ void combine_kernel(const float* __restrict__ eout, const float* __restrict__ gatev,
                               const int* __restrict__ rowid, float* __restrict__ out) {
  const int t = blockIdx.x, tid = threadIdx.x;
  const float g0 = gatev[t * 2], g1 = gatev[t * 2 + 1];
  const float* r0 = eout + (size_t)rowid[t * 2] * DD;
  const float* r1 = eout + (size_t)rowid[t * 2 + 1] * DD;
  for (int n = tid; n < DD; n += 256)
    out[(size_t)t * DD + n] += g0 * r0[n] + g1 * r1[n];
}

// ---------------------------------------------------------------------------
// diag = mean(probs) over tokens (deterministic tree reduction)
// ---------------------------------------------------------------------------
__global__ void diag_kernel(const float* __restrict__ probs, float* __restrict__ dout) {
  __shared__ float red[256];
  const int tid = threadIdx.x, e = tid >> 5, l = tid & 31;
  float s = 0.f;
  for (int t = l; t < NB; t += 32) s += probs[t * EE + e];
  red[tid] = s; __syncthreads();
  for (int st = 16; st > 0; st >>= 1) {
    if (l < st) red[tid] += red[tid + st];
    __syncthreads();
  }
  if (l == 0) dout[e] = red[tid] * (1.0f / NB);
}

// ---------------------------------------------------------------------------
extern "C" void kernel_launch(void* const* d_in, const int* in_sizes, int n_in,
                              void* d_out, int out_size, void* d_ws, size_t ws_size,
                              hipStream_t stream) {
  const float* x     = (const float*)d_in[0];
  const float* ts    = (const float*)d_in[1];
  const float* ln1s  = (const float*)d_in[2];
  const float* ln1b  = (const float*)d_in[3];
  const float* wqkv  = (const float*)d_in[4];
  const float* bqkv  = (const float*)d_in[5];
  const float* wo    = (const float*)d_in[6];
  const float* bo    = (const float*)d_in[7];
  const float* rel   = (const float*)d_in[8];
  const float* ln2s  = (const float*)d_in[9];
  const float* ln2b  = (const float*)d_in[10];
  const float* wtxt  = (const float*)d_in[11];
  const float* btxt  = (const float*)d_in[12];
  const float* wr    = (const float*)d_in[13];
  const float* br    = (const float*)d_in[14];
  const float* w1    = (const float*)d_in[15];
  const float* b1    = (const float*)d_in[16];
  const float* w2    = (const float*)d_in[17];
  const float* b2    = (const float*)d_in[18];
  float* out = (float*)d_out;                 // [B*T*D] x then [E] diag

  // Workspace carve-out (≈92 MB total).
  char* base = (char*)d_ws;
  size_t o = 0;
  auto alloc = [&](size_t bytes) { size_t r = o; o = (o + bytes + 255) & ~(size_t)255; return r; };
  u16*   h1bf   = (u16*)  (base + alloc((size_t)NB * DD * 2));
  u16*   qbf    = (u16*)  (base + alloc((size_t)NB * DD * 2));
  u16*   kbf    = (u16*)  (base + alloc((size_t)NB * DD * 2));
  u16*   vbf    = (u16*)  (base + alloc((size_t)NB * DD * 2));
  u16*   ctxbf  = (u16*)  (base + alloc((size_t)NB * DD * 2));
  u16*   h2bf   = (u16*)  (base + alloc((size_t)NB * DD * 2));
  float* h2f    = (float*)(base + alloc((size_t)NB * DD * 4));
  float* txt    = (float*)(base + alloc((size_t)2 * DD * 4));
  float* probs  = (float*)(base + alloc((size_t)NB * EE * 4));
  float* gatev  = (float*)(base + alloc((size_t)NB * 2 * 4));
  int*   topidx = (int*)  (base + alloc((size_t)NB * 2 * 4));
  int*   cnt    = (int*)  (base + alloc(EE * 4));
  int*   off    = (int*)  (base + alloc((EE + 1) * 4));
  int*   list   = (int*)  (base + alloc((size_t)EE * NB * 4));
  int*   rowid  = (int*)  (base + alloc((size_t)NB * 2 * 4));
  float* scores = (float*)(base + alloc((size_t)32 * TT * TT * 4));
  u16*   attnbf = (u16*)  (base + alloc((size_t)32 * TT * TT * 2));
  u16*   hidbf  = (u16*)  (base + alloc((size_t)2 * NB * FF * 2));
  float* eout   = (float*)(base + alloc((size_t)2 * NB * DD * 4));
  (void)n_in; (void)in_sizes; (void)out_size; (void)ws_size;

  ln_kernel<<<NB, 256, 0, stream>>>(x, ln1s, ln1b, h1bf, nullptr);
  qkv_gemm<<<dim3(48, 16), 128, 0, stream>>>(h1bf, wqkv, bqkv, qbf, kbf, vbf);
  scores_gemm<<<dim3(8, 8, 32), 128, 0, stream>>>(qbf, kbf, rel, scores);
  softmax_kernel<<<32 * TT, 128, 0, stream>>>(scores, attnbf);
  ctx_gemm<<<dim3(8, 32), 128, 0, stream>>>(attnbf, vbf, ctxbf);
  outproj_gemm<<<dim3(16, 16), 128, 0, stream>>>(ctxbf, wo, bo, x, out);
  ln_kernel<<<NB, 256, 0, stream>>>(out, ln2s, ln2b, h2bf, h2f);
  txtproj_kernel<<<16, 128, 0, stream>>>(ts, wtxt, btxt, txt);
  router_kernel<<<NB, 32, 0, stream>>>(h2f, txt, wr, br, probs, gatev, topidx);
  build_lists_kernel<<<1, 32, 0, stream>>>(topidx, cnt, off, list, rowid);
  moe_gemm1<<<dim3(64, 16, 8), 128, 0, stream>>>(h2bf, w1, b1, cnt, off, list, hidbf);
  moe_gemm2<<<dim3(16, 16, 8), 128, 0, stream>>>(hidbf, w2, b2, cnt, off, eout);
  combine_kernel<<<NB, 256, 0, stream>>>(eout, gatev, rowid, out);
  diag_kernel<<<1, 256, 0, stream>>>(probs, out + (size_t)NB * DD);
}
